// LongTermMemory_893353197938
// MI455X (gfx1250) — matrix operand order repaired and checked by reading, not compiled, not run on hardware
//
#include <hip/hip_runtime.h>
#include <hip/hip_bf16.h>
#include <math.h>

typedef __bf16 bf16_t;
typedef __attribute__((ext_vector_type(4)))  __bf16 v4bf;
typedef __attribute__((ext_vector_type(8)))  __bf16 v8bf;
typedef __attribute__((ext_vector_type(16))) __bf16 v16bf;
typedef __attribute__((ext_vector_type(8)))  float  v8f;

#define BQ      1024
#define NMEM    262144
#define DDIM    128
#define TOPK    8
#define NCHUNKS 32
#define CHUNK   (NMEM / NCHUNKS)   // 8192
#define MTILES  (BQ / 32)          // 32 workgroups in x, 32 queries each

#define NEG_BIG (-1.0e30f)

// Insert (v,id) into a descending-sorted top-8 list kept in registers.
// Strict '<=' skip => earliest index wins on ties (matches lax.top_k).
__device__ __forceinline__ void topk_insert(float* s, int* ix, float v, int id) {
    if (v <= s[TOPK - 1]) return;
    s[TOPK - 1] = v;
    ix[TOPK - 1] = id;
#pragma unroll
    for (int j = TOPK - 1; j > 0; --j) {
        if (s[j] > s[j - 1]) {
            float tf = s[j]; s[j] = s[j - 1]; s[j - 1] = tf;
            int   ti = ix[j]; ix[j] = ix[j - 1]; ix[j - 1] = ti;
        }
    }
}

// Scan 8 candidates with a max-tree early-out: in the common case all 8 fail
// the current 8th-best threshold and we pay only 7 v_max + 1 compare.
__device__ __forceinline__ void scan8(float* ts, int* ti,
                                      float4 a, float4 b, int base) {
    const float m = fmaxf(fmaxf(fmaxf(a.x, a.y), fmaxf(a.z, a.w)),
                          fmaxf(fmaxf(b.x, b.y), fmaxf(b.z, b.w)));
    if (m <= ts[TOPK - 1]) return;
    topk_insert(ts, ti, a.x, base + 0);
    topk_insert(ts, ti, a.y, base + 1);
    topk_insert(ts, ti, a.z, base + 2);
    topk_insert(ts, ti, a.w, base + 3);
    topk_insert(ts, ti, b.x, base + 4);
    topk_insert(ts, ti, b.y, base + 5);
    topk_insert(ts, ti, b.z, base + 6);
    topk_insert(ts, ti, b.w, base + 7);
}

// A fragment loader: 16-bit A 16x32 layout. Lanes 0-15 hold K={0..7,16..23},
// lanes 16-31 hold K={8..15,24..31} of each 32-wide K block.
__device__ __forceinline__ void load_afrag(v16bf* a, const bf16_t* qrow, int g) {
#pragma unroll
    for (int kb = 0; kb < 4; ++kb) {
        v8bf lo = *(const v8bf*)(qrow + 32 * kb + 8 * g);
        v8bf hi = *(const v8bf*)(qrow + 32 * kb + 16 + 8 * g);
        a[kb] = __builtin_shufflevector(lo, hi,
                  0, 1, 2, 3, 4, 5, 6, 7, 8, 9, 10, 11, 12, 13, 14, 15);
    }
}

// B fragment loader: 32x16 bf16 — lane holds column n = lane&15; lanes 0-15
// hold K={0..7,16..23}, lanes 16-31 hold K={8..15,24..31} of each 32-wide K block.
__device__ __forceinline__ void load_bfrag(v16bf* b, const bf16_t* mrow, int g) {
#pragma unroll
    for (int kb = 0; kb < 4; ++kb) {
        v8bf lo = *(const v8bf*)(mrow + 32 * kb + 8 * g);
        v8bf hi = *(const v8bf*)(mrow + 32 * kb + 16 + 8 * g);
        b[kb] = __builtin_shufflevector(lo, hi,
                  0, 1, 2, 3, 4, 5, 6, 7, 8, 9, 10, 11, 12, 13, 14, 15);
    }
}

// ---------------- Kernel 1: queries f32 -> bf16 (no normalization needed;
// top-k per query is invariant to positive per-query scaling) ----------------
__global__ __launch_bounds__(256)
void k_convert_q(const float* __restrict__ q, bf16_t* __restrict__ qbf) {
    int i = blockIdx.x * 256 + threadIdx.x;
    qbf[i] = (bf16_t)q[i];
}

// ---------------- Kernel 2: memory rows -> bf16 copy + inverse L2 norms -----
__global__ __launch_bounds__(256)
void k_rownorm(const float* __restrict__ mem, bf16_t* __restrict__ mbf,
               float* __restrict__ invn) {
    const int row  = blockIdx.x * 8 + (threadIdx.x >> 5);   // one wave32 per row
    const int lane = threadIdx.x & 31;
    const float4 v = *(const float4*)(mem + (size_t)row * DDIM + lane * 4);
    float s = v.x * v.x + v.y * v.y + v.z * v.z + v.w * v.w;
#pragma unroll
    for (int m = 16; m > 0; m >>= 1) s += __shfl_xor(s, m, 32);
    v4bf o = { (bf16_t)v.x, (bf16_t)v.y, (bf16_t)v.z, (bf16_t)v.w };
    *(v4bf*)(mbf + (size_t)row * DDIM + lane * 4) = o;
    if (lane == 0) invn[row] = 1.0f / fmaxf(sqrtf(s), 1e-12f);
}

// ---------------- Kernel 3: WMMA GEMM fused with per-chunk top-8 ------------
// grid = (MTILES, NCHUNKS), block = 256 (8 waves). 32 query rows per
// workgroup (two 16-row A tiles reuse every B fragment -> 8 WMMAs per B load).
// Loop unrolled x2 with ping-pong B buffers AND ping-pong invnorm scales
// (consumed value always loaded two bodies earlier -> no loadcnt drain), plus
// double-buffered sims LDS tile (one barrier per iteration instead of two).
__global__ __launch_bounds__(256)
void k_gemm_topk(const bf16_t* __restrict__ qbf, const bf16_t* __restrict__ mbf,
                 const float* __restrict__ invn, float2* __restrict__ cand) {
    __shared__ __align__(16) float sims[2][32][128 + 4];
    __shared__ float cs[32][8][TOPK];
    __shared__ int   ci[32][8][TOPK];

    const int tid  = threadIdx.x;
    const int wave = tid >> 5;
    const int lane = tid & 31;
    const int g    = lane >> 4;    // half-wave (K-split group per ISA layout)
    const int ln   = lane & 15;
    const int qb   = blockIdx.x;
    const int ch   = blockIdx.y;
    const int chunk_base = ch * CHUNK;

    // Two A tiles (query rows qb*32 .. qb*32+31), resident for the whole chunk.
    v16bf a0[4], a1[4];
    const bf16_t* qrow = qbf + (size_t)(qb * 32 + ln) * DDIM;
    load_afrag(a0, qrow, g);
    load_afrag(a1, qrow + 16 * DDIM, g);

    // Per-thread running top-8 stream: query row `selq`, 16 columns per
    // 128-wide iteration starting at `selc`.
    float ts[TOPK]; int ti[TOPK];
#pragma unroll
    for (int j = 0; j < TOPK; ++j) { ts[j] = NEG_BIG; ti[j] = 0; }
    const int selq = tid >> 3;          // 0..31
    const int selc = (tid & 7) * 16;    // 0,16,...,112

    const int iters = CHUNK / 128;                  // 64
    const int col_base = chunk_base + 16 * wave + ln;

    v16bf b0[4], b1[4];
    load_bfrag(b0, mbf + (size_t)col_base * DDIM, g);
    load_bfrag(b1, mbf + (size_t)(col_base + 128) * DDIM, g);
    float s0 = invn[col_base];
    float s1 = invn[col_base + 128];

    auto body = [&](int it, v16bf* bc, float& ss) {
        const int colIt = col_base + it * 128;
        const float scale = ss;                     // loaded two bodies ago

        // 8 WMMAs back-to-back: both A tiles consume the buffered B.
        v8f c0 = {}, c1 = {};
#pragma unroll
        for (int kb = 0; kb < 4; ++kb) {
            c0 = __builtin_amdgcn_wmma_f32_16x16x32_bf16(
                     false, a0[kb], false, bc[kb], (short)0, c0, false, false);
            c1 = __builtin_amdgcn_wmma_f32_16x16x32_bf16(
                     false, a1[kb], false, bc[kb], (short)0, c1, false, false);
        }

        // Branch-free reload of this buffer + its scale for iteration it+2;
        // on the final iterations clamp back to col_base (valid load, result
        // unused) so the outstanding-load count stays path-invariant.
        const int rcol = (it + 2 < iters) ? (colIt + 256) : col_base;
        load_bfrag(bc, mbf + (size_t)rcol * DDIM, g);
        ss = invn[rcol];
        const int pcol = (it + 3 < iters) ? (colIt + 384) : col_base;
        __builtin_prefetch(mbf + (size_t)pcol * DDIM, 0, 3);

        // C layout: VGPR v, lanes 0-15 -> M=v; lanes 16-31 -> M=v+8.
        const int p = it & 1;
#pragma unroll
        for (int v = 0; v < 8; ++v) {
            sims[p][v + 8 * g][16 * wave + ln]      = c0[v] * scale;
            sims[p][16 + v + 8 * g][16 * wave + ln] = c1[v] * scale;
        }
        __syncthreads();   // writes visible; WAR on buffer p is covered by the
                           // intervening body's barrier on buffer 1-p

        {   // four ds_load_b128; max-tree early-out per 8 candidates
            const float4* rp = (const float4*)(&sims[p][selq][selc]);
            const float4 x0 = rp[0];
            const float4 x1 = rp[1];
            const float4 x2 = rp[2];
            const float4 x3 = rp[3];
            const int base = colIt - 16 * wave - ln + selc;   // chunk_base + it*128 + selc
            scan8(ts, ti, x0, x1, base);
            scan8(ts, ti, x2, x3, base + 8);
        }
    };

    for (int it = 0; it < iters; it += 2) {
        body(it, b0, s0);
        body(it + 1, b1, s1);
    }
    __syncthreads();

    // Merge 8 partial top-8 lists per query -> one top-8 per (query, chunk).
#pragma unroll
    for (int j = 0; j < TOPK; ++j) {
        cs[selq][tid & 7][j] = ts[j];
        ci[selq][tid & 7][j] = ti[j];
    }
    __syncthreads();
    if (tid < 32) {
        float fs[TOPK]; int fi[TOPK];
#pragma unroll
        for (int j = 0; j < TOPK; ++j) { fs[j] = NEG_BIG; fi[j] = 0; }
        for (int s = 0; s < 8; ++s)
#pragma unroll
            for (int j = 0; j < TOPK; ++j)
                topk_insert(fs, fi, cs[tid][s][j], ci[tid][s][j]);
        float2* dst = cand + ((size_t)(qb * 32 + tid) * NCHUNKS + ch) * TOPK;
#pragma unroll
        for (int j = 0; j < TOPK; ++j)
            dst[j] = make_float2(fs[j], __int_as_float(fi[j]));
    }
}

// ---------------- Kernel 4: merge chunk candidates + gather rows ------------
__global__ __launch_bounds__(256)
void k_merge_gather(const float2* __restrict__ cand, const float* __restrict__ mem,
                    float* __restrict__ out) {
    __shared__ int idx[TOPK];
    const int q = blockIdx.x;
    if (threadIdx.x == 0) {
        float fs[TOPK]; int fi[TOPK];
#pragma unroll
        for (int j = 0; j < TOPK; ++j) { fs[j] = NEG_BIG; fi[j] = 0; }
        const float2* c = cand + (size_t)q * NCHUNKS * TOPK;
        for (int j = 0; j < NCHUNKS * TOPK; ++j)
            topk_insert(fs, fi, c[j].x, __float_as_int(c[j].y));
#pragma unroll
        for (int j = 0; j < TOPK; ++j) idx[j] = fi[j];
    }
    __syncthreads();
    for (int e = threadIdx.x; e < TOPK * DDIM; e += 256) {
        const int r = e >> 7, cidx = e & 127;
        out[(size_t)q * TOPK * DDIM + e] = mem[(size_t)idx[r] * DDIM + cidx];
    }
}

extern "C" void kernel_launch(void* const* d_in, const int* in_sizes, int n_in,
                              void* d_out, int out_size, void* d_ws, size_t ws_size,
                              hipStream_t stream) {
    (void)in_sizes; (void)n_in; (void)out_size; (void)ws_size;
    const float* q   = (const float*)d_in[0];   // [B, D] f32
    const float* mem = (const float*)d_in[1];   // [N, D] f32
    // d_in[2] = top_k scalar (fixed 8, baked into TOPK)
    float* out = (float*)d_out;                 // [B, TOPK, D] f32

    char* ws = (char*)d_ws;
    const size_t off_mbf  = 0;                                   // N*D*2 = 64 MB
    const size_t off_invn = off_mbf  + (size_t)NMEM * DDIM * 2;  // N*4   =  1 MB
    const size_t off_qbf  = off_invn + (size_t)NMEM * 4;         // B*D*2 = 256 KB
    const size_t off_cand = off_qbf  + (size_t)BQ * DDIM * 2;    // B*32*8*8 = 2 MB
    bf16_t* mbf  = (bf16_t*)(ws + off_mbf);
    float*  invn = (float*)(ws + off_invn);
    bf16_t* qbf  = (bf16_t*)(ws + off_qbf);
    float2* cand = (float2*)(ws + off_cand);

    k_convert_q<<<(BQ * DDIM) / 256, 256, 0, stream>>>(q, qbf);
    k_rownorm<<<NMEM / 8, 256, 0, stream>>>(mem, mbf, invn);
    k_gemm_topk<<<dim3(MTILES, NCHUNKS), 256, 0, stream>>>(qbf, mbf, invn, cand);
    k_merge_gather<<<BQ, 256, 0, stream>>>(cand, mem, out);
}
